// LSTM_472446402732
// MI455X (gfx1250) — matrix operand order repaired and checked by reading, not compiled, not run on hardware
//
#include <hip/hip_runtime.h>

// ---------------------------------------------------------------------------
// Fused LSTM for MI455X (gfx1250, wave32, WMMA).  B=256,T=1024,I=64,H=128.
// 16 WGs x 256 threads (8 waves). Each WG owns 16 batch rows, runs the full
// sequential T loop; W_ih/W_hh resident in LDS (bf16); 16 B-fragments
// hoisted into registers (W_hh gates i,f,g + W_ih gates i,f). x and h tiles
// are double-buffered (unswizzled) in LDS so one contiguous 32B load gives the
// WMMA A-fragment and only ONE barrier per timestep is needed.
// Gate accumulators start from the WMMA inline-0 C operand; biases are added
// in the activation epilogue (fused into the exp2 argument as FMAs).
// Per step per wave: 24 x v_wmma_f32_16x16x32_bf16.
// ---------------------------------------------------------------------------

typedef __attribute__((ext_vector_type(16))) __bf16 v16bf;
typedef __attribute__((ext_vector_type(4)))  __bf16 bf16x4;
typedef __attribute__((ext_vector_type(8)))  float  v8f;

#define B_TOT 256
#define T_LEN 1024
#define I_DIM 64
#define H_DIM 128
#define G_DIM 512  // 4*H
#define ROWS  16   // batch rows per workgroup

// LDS layout (bytes)
#define LDS_WIH   0                         // [512][64]  bf16 = 65536
#define LDS_WHH   (LDS_WIH + G_DIM*I_DIM*2) // [512][128] bf16 = 131072
#define LDS_BIAS  (LDS_WHH + G_DIM*H_DIM*2) // [512]      f32  = 2048
#define LDS_H     (LDS_BIAS + G_DIM*4)      // [2][16][128] bf16 = 8192 (unswizzled)
#define LDS_X     (LDS_H + 2*ROWS*H_DIM*2)  // [2][16][64]  bf16 = 4096 (unswizzled)
#define LDS_TOTAL (LDS_X + 2*ROWS*I_DIM*2)  // = 210944

// Swizzle for A-operand tiles: swap bits 3<->4 of the K index (within each
// 32-wide K chunk).  Reading 16 contiguous elements at [row][kc*32 + hi*16]
// then yields exactly the 16-bit WMMA A fragment:
//   elements 0..7  -> K = kc*32 + 8*hi + 0..7
//   elements 8..15 -> K = kc*32 + 8*hi + 16 + 0..7
static __device__ __forceinline__ int swz(int k) {
    return (k & ~24) | ((k & 8) << 1) | ((k & 16) >> 1);
}

static __device__ __forceinline__ float sigmoid_f(float x) {
    return 1.0f / (1.0f + __expf(-x));
}
static __device__ __forceinline__ float tanh_f(float x) {
    float e = __expf(2.0f * x);
    return (e - 1.0f) / (e + 1.0f);
}

__global__ __launch_bounds__(256, 2)
void lstm_fused_kernel(const float* __restrict__ data,   // [B][T][I]
                       const float* __restrict__ W_ih,   // [4H][I]
                       const float* __restrict__ W_hh,   // [4H][H]
                       const float* __restrict__ b_ih,   // [4H]
                       const float* __restrict__ b_hh,   // [4H]
                       float* __restrict__ out) {        // [B][T][H]
    extern __shared__ __align__(16) char smem[];
    __bf16* wih  = (__bf16*)(smem + LDS_WIH);
    __bf16* whh  = (__bf16*)(smem + LDS_WHH);
    float*  bias = (float*) (smem + LDS_BIAS);
    __bf16* hsm  = (__bf16*)(smem + LDS_H);   // double buffered
    __bf16* xsm  = (__bf16*)(smem + LDS_X);   // double buffered

    const int tid   = threadIdx.x;
    const int w     = tid >> 5;          // wave id 0..7 -> hidden cols 16w..16w+15
    const int lane  = tid & 31;
    const int ln    = lane & 15;         // N column of B/D, row M of A
    const int hi    = lane >> 4;         // lane half
    const int bbase = blockIdx.x * ROWS;

    // ---- prologue: stage weights (f32->bf16), bias sum, zero h buf0 ----
    for (int i = tid; i < G_DIM * I_DIM; i += 256) wih[i] = (__bf16)W_ih[i];
    for (int i = tid; i < G_DIM * H_DIM; i += 256) whh[i] = (__bf16)W_hh[i];
    for (int i = tid; i < G_DIM; i += 256)         bias[i] = b_ih[i] + b_hh[i];
    for (int i = tid; i < ROWS * H_DIM; i += 256)  hsm[i] = (__bf16)0.0f;

    // x tile t=0 into buffer 0 (unswizzled store)
    const int xr    = tid >> 4;
    const int xk4   = (tid & 15) * 4;
    const int xk4s  = xk4;               // unswizzled store index for x
    {
        const float4 x0 = *(const float4*)(data + (size_t)(bbase + xr) * T_LEN * I_DIM + xk4);
        bf16x4 v; v[0] = (__bf16)x0.x; v[1] = (__bf16)x0.y;
                  v[2] = (__bf16)x0.z; v[3] = (__bf16)x0.w;
        *(bf16x4*)(xsm + xr * I_DIM + xk4s) = v;
    }
    __syncthreads();

    // ---- hoisted B-fragments ----
    // B fragment (32x16): lane ln = column N, elements 0..15 = K 16*hi+0..15,
    // i.e. 32 contiguous bytes of the [n][k] row-major weight row.
    v16bf Bh[3][4];                      // W_hh gates 0..2 (i,f,g)
#pragma unroll
    for (int g = 0; g < 3; ++g) {
        const int n = g * H_DIM + w * 16 + ln;
#pragma unroll
        for (int kc = 0; kc < 4; ++kc)
            Bh[g][kc] = *(const v16bf*)(whh + n * H_DIM + kc * 32 + hi * 16);
    }
    v16bf Bx[2][2];                      // W_ih gates 0..1 (i,f)
#pragma unroll
    for (int g = 0; g < 2; ++g) {
        const int n = g * H_DIM + w * 16 + ln;
#pragma unroll
        for (int kc = 0; kc < 2; ++kc)
            Bx[g][kc] = *(const v16bf*)(wih + n * I_DIM + kc * 32 + hi * 16);
    }
    float bias_g[4];
#pragma unroll
    for (int g = 0; g < 4; ++g) bias_g[g] = bias[g * H_DIM + w * 16 + ln];

    const int hcol  = w * 16 + ln;       // hidden column owned by this lane
    const int hcols = hcol;              // unswizzled store index for h
    v8f c_st = {};                       // cell state: elem e -> (row e+8*hi, col hcol)

#pragma unroll 1
    for (int t = 0; t < T_LEN; ++t) {
        // prefetch next x tile (global -> regs), hidden behind the WMMA chain
        float4 xpre = make_float4(0.f, 0.f, 0.f, 0.f);
        if (t + 1 < T_LEN) {
            xpre = *(const float4*)(data + (size_t)(bbase + xr) * T_LEN * I_DIM
                                         + (size_t)(t + 1) * I_DIM + xk4);
        }

        // A operands: single contiguous 32B loads from unswizzled LDS buffers
        const __bf16* xb = xsm + (t & 1) * ROWS * I_DIM;
        const __bf16* hb = hsm + (t & 1) * ROWS * H_DIM;
        v16bf Ax[2], Ah[4];
#pragma unroll
        for (int kc = 0; kc < 2; ++kc)
            Ax[kc] = *(const v16bf*)(xb + ln * I_DIM + kc * 32 + hi * 16);
#pragma unroll
        for (int kc = 0; kc < 4; ++kc)
            Ah[kc] = *(const v16bf*)(hb + ln * H_DIM + kc * 32 + hi * 16);

        // gates = x_t @ W_ih^T + h @ W_hh^T  (bias added in epilogue).
        // Each chain starts from the WMMA inline-0 C operand.
        v8f acc[4];
        // x-projection: gates 0,1 from registers; gates 2,3 from LDS
#pragma unroll
        for (int g = 0; g < 2; ++g) {
            v8f a = {};
#pragma unroll
            for (int kc = 0; kc < 2; ++kc)
                a = __builtin_amdgcn_wmma_f32_16x16x32_bf16(false, Ax[kc], false, Bx[g][kc],
                                                            (short)0, a, false, false);
            acc[g] = a;
        }
#pragma unroll
        for (int g = 2; g < 4; ++g) {
            const int n = g * H_DIM + w * 16 + ln;
            v8f a = {};
#pragma unroll
            for (int kc = 0; kc < 2; ++kc) {
                v16bf bx = *(const v16bf*)(wih + n * I_DIM + kc * 32 + hi * 16);
                a = __builtin_amdgcn_wmma_f32_16x16x32_bf16(false, Ax[kc], false, bx,
                                                            (short)0, a, false, false);
            }
            acc[g] = a;
        }
        // recurrent part: gates 0..2 from registers; gate 3 from LDS
#pragma unroll
        for (int g = 0; g < 3; ++g)
#pragma unroll
            for (int kc = 0; kc < 4; ++kc)
                acc[g] = __builtin_amdgcn_wmma_f32_16x16x32_bf16(false, Ah[kc], false, Bh[g][kc],
                                                                 (short)0, acc[g], false, false);
        {
            const int n = 3 * H_DIM + w * 16 + ln;
#pragma unroll
            for (int kc = 0; kc < 4; ++kc) {
                v16bf bo = *(const v16bf*)(whh + n * H_DIM + kc * 32 + hi * 16);
                acc[3] = __builtin_amdgcn_wmma_f32_16x16x32_bf16(false, Ah[kc], false, bo,
                                                                 (short)0, acc[3], false, false);
            }
        }

        // elementwise LSTM cell update (bias folded into activation FMAs)
        v8f hval;
#pragma unroll
        for (int e = 0; e < 8; ++e) {
            const float ig = sigmoid_f(acc[0][e] + bias_g[0]);
            const float fg = sigmoid_f(acc[1][e] + bias_g[1]);
            const float gg = tanh_f(acc[2][e] + bias_g[2]);
            const float og = sigmoid_f(acc[3][e] + bias_g[3]);
            const float c  = fg * c_st[e] + ig * gg;
            c_st[e] = c;
            hval[e] = og * tanh_f(c);
        }

        // store h_t to global output [B][T][H]
#pragma unroll
        for (int e = 0; e < 8; ++e) {
            const int m = e + hi * 8;
            out[(size_t)(bbase + m) * T_LEN * H_DIM + (size_t)t * H_DIM + hcol] = hval[e];
        }

        // publish h (bf16, unswizzled) into the OTHER buffer, plus the
        // prefetched x tile; then one barrier.
        __bf16* hb_n = hsm + ((t + 1) & 1) * ROWS * H_DIM;
#pragma unroll
        for (int e = 0; e < 8; ++e) {
            const int m = e + hi * 8;
            hb_n[m * H_DIM + hcols] = (__bf16)hval[e];
        }
        if (t + 1 < T_LEN) {
            bf16x4 v; v[0] = (__bf16)xpre.x; v[1] = (__bf16)xpre.y;
                      v[2] = (__bf16)xpre.z; v[3] = (__bf16)xpre.w;
            *(bf16x4*)(xsm + ((t + 1) & 1) * ROWS * I_DIM + xr * I_DIM + xk4s) = v;
        }
        __syncthreads();
    }
}

extern "C" void kernel_launch(void* const* d_in, const int* in_sizes, int n_in,
                              void* d_out, int out_size, void* d_ws, size_t ws_size,
                              hipStream_t stream) {
    (void)in_sizes; (void)n_in; (void)out_size; (void)d_ws; (void)ws_size;
    const float* data = (const float*)d_in[0];
    const float* W_ih = (const float*)d_in[1];
    const float* W_hh = (const float*)d_in[2];
    const float* b_ih = (const float*)d_in[3];
    const float* b_hh = (const float*)d_in[4];
    float* out = (float*)d_out;

    dim3 grid(B_TOT / ROWS);   // 16 workgroups
    dim3 block(256);           // 8 waves (wave32)
    lstm_fused_kernel<<<grid, block, LDS_TOTAL, stream>>>(data, W_ih, W_hh, b_ih, b_hh, out);
}
